// TGN_21492016349928
// MI455X (gfx1250) — compile-verified, matrix operand order checked by reading, not run on hardware
//
#include <hip/hip_runtime.h>
#include <math.h>

typedef __attribute__((ext_vector_type(16))) __bf16 v16bf;
typedef __attribute__((ext_vector_type(8)))  float  v8f;
typedef unsigned short u16;

// ---- problem constants (match reference) ----
#define NN 50000
#define EE 200000
#define FF 128
#define MM 128
#define FEAT_E 64
#define HH 256
#define TT 16
#define OO 64
#define NHEAD 4
#define HD 64
#define KM_PAD 352   // msg K: 256+64+16 -> pad to 11*32
#define KQ_PAD 288   // q_in K: 256+16  -> pad to 9*32
#define KT_M 11
#define KT_Q 9

__device__ __forceinline__ u16 f2bf(float f) {
    unsigned u = __builtin_bit_cast(unsigned, f);
    unsigned r = (u + 0x7FFFu + ((u >> 16) & 1u)) >> 16;
    return (u16)r;
}

struct BF16x16Raw { uint4 lo; uint4 hi; };

// A fragment: 16x32 bf16 tile staged row-major in LDS with row stride `ld` (elems).
__device__ __forceinline__ v16bf load_a_frag(const u16* base, int ld, int lane, int kt) {
    int r  = lane & 15;
    int kb = kt * 32 + ((lane & 16) ? 8 : 0);
    const u16* p = base + r * ld + kb;
    BF16x16Raw f;
    f.lo = *(const uint4*)(p);
    f.hi = *(const uint4*)(p + 16);
    return __builtin_bit_cast(v16bf, f);
}

// B fragment: pre-packed tiles of 512 bf16; lane-contiguous.
__device__ __forceinline__ v16bf load_b_frag(const u16* tiles, int tile, int lane) {
    return *(const v16bf*)(tiles + ((size_t)tile << 9) + lane * 16);
}

__device__ __forceinline__ v8f wmma_bf16(v16bf a, v16bf b, v8f c) {
    return __builtin_amdgcn_wmma_f32_16x16x32_bf16(false, a, false, b, (short)0, c, false, false);
}

// ------------------------------------------------------------------
// Weight packing: fp32 [out_dim x in_dim] -> bf16 WMMA B-tiles.
// tile index = n_tile * k_tiles + k_tile ; within tile: lane*16 + j
// lane holds col n = nt*16 + (lane&15); K pattern mirrors A layout.
// ------------------------------------------------------------------
__global__ void pack_w_kernel(const float* __restrict__ src, u16* __restrict__ dst,
                              int out_dim, int in_dim, int k_tiles, int total) {
    int idx = blockIdx.x * 256 + threadIdx.x;
    if (idx >= total) return;
    int tile = idx >> 9, within = idx & 511;
    int lane = within >> 4, j = within & 15;
    int nt = tile / k_tiles, kt = tile - nt * k_tiles;
    int n  = nt * 16 + (lane & 15);
    int kb = kt * 32 + ((lane & 16) ? 8 : 0);
    int k  = kb + ((j < 8) ? j : (8 + j));
    float v = (n < out_dim && k < in_dim) ? src[(size_t)n * in_dim + k] : 0.f;
    dst[idx] = f2bf(v);
}

// q_te (query-time encoding at t=0) = [phi0, sin(phi1..15)], per layer, padded to 32.
__global__ void qte_kernel(const float* __restrict__ te_phi, u16* __restrict__ qte) {
    int t = threadIdx.x;            // 64 threads
    if (t >= 64) return;
    int l = t >> 5, c = t & 31;
    float v = 0.f;
    if (c < TT) {
        float wt = te_phi[l * TT + c];
        v = (c == 0) ? wt : sinf(wt);
    }
    qte[l * 32 + c] = f2bf(v);
}

__global__ void zero_f32_kernel(float* __restrict__ p, int n) {
    int i = blockIdx.x * 256 + threadIdx.x;
    if (i < n) p[i] = 0.f;
}

// ------------------------------------------------------------------
// Input projection: h = [node_feat|memory] @ in_w.T + in_b
// Block: 16 rows x 256 cols, 8 waves x 2 col-tiles, K = 256 (8 k-tiles)
// ------------------------------------------------------------------
__global__ __launch_bounds__(256) void in_proj_kernel(
    const float* __restrict__ nodef, const float* __restrict__ mem,
    const u16* __restrict__ w_p, const float* __restrict__ bias,
    float* __restrict__ h, u16* __restrict__ hb) {
    __shared__ __align__(16) u16 s_a[16 * 256];
    int t = threadIdx.x;
    int n0 = blockIdx.x * 16;
    {
        int e = t >> 4, seg = t & 15;
        int n = n0 + e;
        const float* sp = (seg < 8) ? (nodef + (size_t)n * FF + seg * 16)
                                    : (mem   + (size_t)n * MM + (seg - 8) * 16);
        u16* dp = s_a + e * 256 + seg * 16;
#pragma unroll
        for (int i = 0; i < 16; ++i) dp[i] = f2bf(sp[i]);
    }
    __syncthreads();
    int wave = t >> 5, lane = t & 31, hf = lane >> 4;
    v8f acc[2] = {v8f{}, v8f{}};
#pragma unroll
    for (int kt = 0; kt < 8; ++kt) {
        v16bf a = load_a_frag(s_a, 256, lane, kt);
#pragma unroll
        for (int c = 0; c < 2; ++c) {
            int ct = wave * 2 + c;
            v16bf b = load_b_frag(w_p, ct * 8 + kt, lane);
            acc[c] = wmma_bf16(a, b, acc[c]);
        }
    }
#pragma unroll
    for (int c = 0; c < 2; ++c) {
        int col = (wave * 2 + c) * 16 + (lane & 15);
        float bv = bias[col];
#pragma unroll
        for (int j = 0; j < 8; ++j) {
            int r = j + 8 * hf;
            float v = acc[c][j] + bv;
            size_t o = (size_t)(n0 + r) * HH + col;
            h[o]  = v;
            hb[o] = f2bf(v);
        }
    }
}

// ------------------------------------------------------------------
// Fused edge attention kernel (one layer). Block = 16 edges, 8 waves.
// ------------------------------------------------------------------
__global__ __launch_bounds__(256) void edge_attn_kernel(
    const u16*  __restrict__ hb,
    const int*  __restrict__ edge_index,
    const float* __restrict__ edge_feat,
    const float* __restrict__ edge_times,
    const float* __restrict__ te_w,   // layer slice (16)
    const float* __restrict__ te_phi, // layer slice (16)
    const u16*  __restrict__ wq_p, const float* __restrict__ wq_b,
    const u16*  __restrict__ wk_p, const float* __restrict__ wk_b,
    const u16*  __restrict__ wv_p, const float* __restrict__ wv_b,
    const u16*  __restrict__ qte,    // 32 bf16
    float* __restrict__ agg, float* __restrict__ nrm) {
    __shared__ __align__(16) u16 s_msg[16 * KM_PAD];
    __shared__ __align__(16) u16 s_qin[16 * KQ_PAD];
    __shared__ __align__(16) float s_scores[16 * NHEAD];
    __shared__ __align__(16) float s_af[16];
    __shared__ int s_src[16], s_dst[16], s_dd[16];
    __shared__ float s_td[16];

    int t = threadIdx.x;
    int e0 = blockIdx.x * 16;

    if (t < 16) {
        int e = e0 + t;
        int s = edge_index[e];
        int d = edge_index[EE + e];
        s_src[t] = s;
        s_dst[t] = d;
        s_dd[t]  = edge_index[EE + d];   // dst[dst[e]]
        s_td[t]  = 0.f - edge_times[e];
    }
    if (t < 64) s_scores[t] = 0.f;
    __syncthreads();

    {   // stage gathered h rows (bf16) into msg/q_in
        int e = t >> 4, seg = t & 15;
        const uint4* sp = (const uint4*)(hb + (size_t)s_src[e] * HH + seg * 16);
        uint4 a0 = sp[0], a1 = sp[1];
        uint4* dp = (uint4*)(s_msg + e * KM_PAD + seg * 16);
        dp[0] = a0; dp[1] = a1;
        const uint4* qp = (const uint4*)(hb + (size_t)s_dd[e] * HH + seg * 16);
        uint4 b0 = qp[0], b1 = qp[1];
        uint4* dq = (uint4*)(s_qin + e * KQ_PAD + seg * 16);
        dq[0] = b0; dq[1] = b1;
    }
    if (t < 64) {  // edge_feat (64 f32 -> bf16)
        int e = t >> 2, seg = t & 3;
        const float* fp = edge_feat + (size_t)(e0 + e) * FEAT_E + seg * 16;
        u16* dp = s_msg + e * KM_PAD + HH + seg * 16;
#pragma unroll
        for (int i = 0; i < 16; ++i) dp[i] = f2bf(fp[i]);
    }
    {   // time encodings + padding + q_te
        int e = t >> 4, c = t & 15;
        float wt = s_td[e] * te_w[c] + te_phi[c];
        float tv = (c == 0) ? wt : sinf(wt);
        s_msg[e * KM_PAD + HH + FEAT_E + c] = f2bf(tv);
        s_msg[e * KM_PAD + HH + FEAT_E + TT + c] = 0;
        s_qin[e * KQ_PAD + HH + c] = qte[c];
        s_qin[e * KQ_PAD + HH + TT + c] = 0;
    }
    __syncthreads();

    int wave = t >> 5, lane = t & 31, hf = lane >> 4;
    v8f accQ[2] = {v8f{}, v8f{}};
    v8f accK[2] = {v8f{}, v8f{}};
    v8f accV[2] = {v8f{}, v8f{}};

#pragma unroll
    for (int kt = 0; kt < KT_Q; ++kt) {
        v16bf a = load_a_frag(s_qin, KQ_PAD, lane, kt);
#pragma unroll
        for (int c = 0; c < 2; ++c) {
            int ct = wave * 2 + c;
            v16bf b = load_b_frag(wq_p, ct * KT_Q + kt, lane);
            accQ[c] = wmma_bf16(a, b, accQ[c]);
        }
    }
#pragma unroll
    for (int kt = 0; kt < KT_M; ++kt) {
        v16bf a = load_a_frag(s_msg, KM_PAD, lane, kt);
#pragma unroll
        for (int c = 0; c < 2; ++c) {
            int ct = wave * 2 + c;
            v16bf bk = load_b_frag(wk_p, ct * KT_M + kt, lane);
            accK[c] = wmma_bf16(a, bk, accK[c]);
            v16bf bv = load_b_frag(wv_p, ct * KT_M + kt, lane);
            accV[c] = wmma_bf16(a, bv, accV[c]);
        }
    }

    // partial scores: per-lane Q*K products, butterfly-reduce across 16 column lanes
#pragma unroll
    for (int c = 0; c < 2; ++c) {
        int ct = wave * 2 + c;
        int col = ct * 16 + (lane & 15);
        float bq = wq_b[col], bk = wk_b[col], bv = wv_b[col];
        int head = ct >> 2;   // 4 col-tiles per head, uniform per tile
#pragma unroll
        for (int j = 0; j < 8; ++j) {
            float prod = (accQ[c][j] + bq) * (accK[c][j] + bk);
            prod += __shfl_xor(prod, 1, 32);
            prod += __shfl_xor(prod, 2, 32);
            prod += __shfl_xor(prod, 4, 32);
            prod += __shfl_xor(prod, 8, 32);
            if ((lane & 15) == 0)
                atomicAdd(&s_scores[(j + 8 * hf) * NHEAD + head], prod);
            accV[c][j] += bv;
        }
    }
    __syncthreads();

    if (t < 16) {
        float af = 0.f;
#pragma unroll
        for (int hh = 0; hh < NHEAD; ++hh) {
            float sc = s_scores[t * NHEAD + hh] * 0.125f;   // / sqrt(64)
            sc = fminf(fmaxf(sc, -5.f), 5.f);
            af += expf(sc);
        }
        af *= 0.25f;
        s_af[t] = af;
        atomicAdd(&nrm[s_dst[t]], af);
    }
    __syncthreads();

    float afv[8];
    {
        const float4* ap = (const float4*)(s_af + hf * 8);
        float4 a0 = ap[0], a1 = ap[1];
        afv[0]=a0.x; afv[1]=a0.y; afv[2]=a0.z; afv[3]=a0.w;
        afv[4]=a1.x; afv[5]=a1.y; afv[6]=a1.z; afv[7]=a1.w;
    }
#pragma unroll
    for (int c = 0; c < 2; ++c) {
        int col = (wave * 2 + c) * 16 + (lane & 15);
#pragma unroll
        for (int j = 0; j < 8; ++j) {
            int r = j + 8 * hf;
            atomicAdd(&agg[(size_t)s_dst[r] * HH + col], accV[c][j] * afv[j]);
        }
    }
}

// ------------------------------------------------------------------
// Node update: h_new = LN1(agg/(nrm+eps)); h = LN2(h + h_new)
// One wave per node (8 elems/lane), shuffle reductions.
// ------------------------------------------------------------------
__device__ __forceinline__ float wave_sum(float s) {
    s += __shfl_xor(s, 1, 32);
    s += __shfl_xor(s, 2, 32);
    s += __shfl_xor(s, 4, 32);
    s += __shfl_xor(s, 8, 32);
    s += __shfl_xor(s, 16, 32);
    return s;
}

__global__ __launch_bounds__(256) void node_update_kernel(
    const float* __restrict__ agg, const float* __restrict__ nrm,
    const float* __restrict__ g1, const float* __restrict__ b1,
    const float* __restrict__ g2, const float* __restrict__ b2,
    float* __restrict__ h, u16* __restrict__ hb) {
    int wave = threadIdx.x >> 5, lane = threadIdx.x & 31;
    int n = blockIdx.x * 8 + wave;
    float inv = 1.f / (nrm[n] + 1e-8f);
    float x[8];
    {
        const float4* p = (const float4*)(agg + (size_t)n * HH + lane * 8);
        float4 a0 = p[0], a1 = p[1];
        x[0]=a0.x*inv; x[1]=a0.y*inv; x[2]=a0.z*inv; x[3]=a0.w*inv;
        x[4]=a1.x*inv; x[5]=a1.y*inv; x[6]=a1.z*inv; x[7]=a1.w*inv;
    }
    float s = 0.f;
#pragma unroll
    for (int j = 0; j < 8; ++j) s += x[j];
    float mean = wave_sum(s) * (1.f / HH);
    float v = 0.f;
#pragma unroll
    for (int j = 0; j < 8; ++j) { float d = x[j] - mean; v += d * d; }
    float rstd = rsqrtf(wave_sum(v) * (1.f / HH) + 1e-5f);
    float y[8];
    {
        const float4* p = (const float4*)(h + (size_t)n * HH + lane * 8);
        float4 a0 = p[0], a1 = p[1];
        float hold[8] = {a0.x,a0.y,a0.z,a0.w,a1.x,a1.y,a1.z,a1.w};
#pragma unroll
        for (int j = 0; j < 8; ++j) {
            int c = lane * 8 + j;
            float hn = (x[j] - mean) * rstd * g1[c] + b1[c];
            y[j] = hold[j] + hn;
        }
    }
    float s2 = 0.f;
#pragma unroll
    for (int j = 0; j < 8; ++j) s2 += y[j];
    float mean2 = wave_sum(s2) * (1.f / HH);
    float v2 = 0.f;
#pragma unroll
    for (int j = 0; j < 8; ++j) { float d = y[j] - mean2; v2 += d * d; }
    float rstd2 = rsqrtf(wave_sum(v2) * (1.f / HH) + 1e-5f);
#pragma unroll
    for (int j = 0; j < 8; ++j) {
        int c = lane * 8 + j;
        float o = (y[j] - mean2) * rstd2 * g2[c] + b2[c];
        size_t idx = (size_t)n * HH + c;
        h[idx]  = o;
        hb[idx] = f2bf(o);
    }
}

// ------------------------------------------------------------------
// Output: out = gelu(h @ out_w.T + out_b).  Block: 32 rows x 64 cols.
// ------------------------------------------------------------------
__global__ __launch_bounds__(256) void out_proj_kernel(
    const u16* __restrict__ hb, const u16* __restrict__ w_p,
    const float* __restrict__ bias, float* __restrict__ out) {
    __shared__ __align__(16) u16 s_a[32 * 256];
    int t = threadIdx.x;
    int n0 = blockIdx.x * 32;
#pragma unroll
    for (int rep = 0; rep < 2; ++rep) {
        int chunk = t + rep * 256;
        int e = chunk >> 4, seg = chunk & 15;
        int n = n0 + e;
        uint4* dp = (uint4*)(s_a + e * 256 + seg * 16);
        if (n < NN) {
            const uint4* sp = (const uint4*)(hb + (size_t)n * HH + seg * 16);
            dp[0] = sp[0]; dp[1] = sp[1];
        } else {
            uint4 z = {0, 0, 0, 0};
            dp[0] = z; dp[1] = z;
        }
    }
    __syncthreads();
    int wave = t >> 5, lane = t & 31, hf = lane >> 4;
    int rt = wave >> 2, ct = wave & 3;
    v8f acc = v8f{};
#pragma unroll
    for (int kt = 0; kt < 8; ++kt) {
        v16bf a = load_a_frag(s_a + rt * 16 * 256, 256, lane, kt);
        v16bf b = load_b_frag(w_p, ct * 8 + kt, lane);
        acc = wmma_bf16(a, b, acc);
    }
    int col = ct * 16 + (lane & 15);
    float bv = bias[col];
#pragma unroll
    for (int j = 0; j < 8; ++j) {
        int n = n0 + rt * 16 + j + 8 * hf;
        if (n < NN) {
            float x = acc[j] + bv;
            float g = 0.5f * x * (1.f + erff(x * 0.70710678118654752f));
            out[(size_t)n * OO + col] = g;
        }
    }
}

// ------------------------------------------------------------------
extern "C" void kernel_launch(void* const* d_in, const int* in_sizes, int n_in,
                              void* d_out, int out_size, void* d_ws, size_t ws_size,
                              hipStream_t stream) {
    (void)in_sizes; (void)n_in; (void)out_size; (void)ws_size;
    const float* node_feat  = (const float*)d_in[0];
    const float* memory     = (const float*)d_in[1];
    const int*   edge_index = (const int*)d_in[2];
    const float* edge_feat  = (const float*)d_in[3];
    const float* edge_times = (const float*)d_in[4];
    const float* in_w   = (const float*)d_in[5];
    const float* in_b   = (const float*)d_in[6];
    const float* te_w   = (const float*)d_in[7];
    const float* te_phi = (const float*)d_in[8];
    const float* wq_w   = (const float*)d_in[9];
    const float* wq_b   = (const float*)d_in[10];
    const float* wk_w   = (const float*)d_in[11];
    const float* wk_b   = (const float*)d_in[12];
    const float* wv_w   = (const float*)d_in[13];
    const float* wv_b   = (const float*)d_in[14];
    const float* aln_w  = (const float*)d_in[15];
    const float* aln_b  = (const float*)d_in[16];
    const float* ln_w   = (const float*)d_in[17];
    const float* ln_b   = (const float*)d_in[18];
    const float* out_w  = (const float*)d_in[19];
    const float* out_b  = (const float*)d_in[20];

    // workspace carve-up (256B aligned)
    char* ws = (char*)d_ws;
    size_t off = 0;
    auto carve = [&](size_t bytes) -> void* {
        void* p = ws + off;
        off = (off + bytes + 255) & ~(size_t)255;
        return p;
    };
    float* h    = (float*)carve((size_t)NN * HH * 4);
    float* agg  = (float*)carve((size_t)NN * HH * 4);
    float* nrm  = (float*)carve((size_t)NN * 4);
    u16*   hb   = (u16*)carve((size_t)NN * HH * 2);
    const int IN_TILES  = 16 * 8;            // 256x256
    const int Q_TILES   = 16 * KT_Q;         // 256x288
    const int M_TILES   = 16 * KT_M;         // 256x352
    const int OUT_TILES = 4 * 8;             // 64x256
    u16* w_in_p  = (u16*)carve((size_t)IN_TILES * 512 * 2);
    u16* wq_p    = (u16*)carve((size_t)2 * Q_TILES * 512 * 2);
    u16* wk_p    = (u16*)carve((size_t)2 * M_TILES * 512 * 2);
    u16* wv_p    = (u16*)carve((size_t)2 * M_TILES * 512 * 2);
    u16* w_out_p = (u16*)carve((size_t)OUT_TILES * 512 * 2);
    u16* qte     = (u16*)carve(2 * 32 * 2);

    // 1) pack weights to WMMA bf16 tiles
    {
        int tot = IN_TILES * 512;
        pack_w_kernel<<<(tot + 255) / 256, 256, 0, stream>>>(in_w, w_in_p, HH, FF + MM, 8, tot);
    }
    for (int l = 0; l < 2; ++l) {
        int totq = Q_TILES * 512;
        pack_w_kernel<<<(totq + 255) / 256, 256, 0, stream>>>(
            wq_w + (size_t)l * HH * (HH + TT), wq_p + (size_t)l * Q_TILES * 512,
            HH, HH + TT, KT_Q, totq);
        int totm = M_TILES * 512;
        pack_w_kernel<<<(totm + 255) / 256, 256, 0, stream>>>(
            wk_w + (size_t)l * HH * (HH + FEAT_E + TT), wk_p + (size_t)l * M_TILES * 512,
            HH, HH + FEAT_E + TT, KT_M, totm);
        pack_w_kernel<<<(totm + 255) / 256, 256, 0, stream>>>(
            wv_w + (size_t)l * HH * (HH + FEAT_E + TT), wv_p + (size_t)l * M_TILES * 512,
            HH, HH + FEAT_E + TT, KT_M, totm);
    }
    {
        int tot = OUT_TILES * 512;
        pack_w_kernel<<<(tot + 255) / 256, 256, 0, stream>>>(out_w, w_out_p, OO, HH, 8, tot);
    }
    qte_kernel<<<1, 64, 0, stream>>>(te_phi, qte);

    // 2) input projection
    in_proj_kernel<<<NN / 16, 256, 0, stream>>>(node_feat, memory, w_in_p, in_b, h, hb);

    // 3) attention layers
    for (int l = 0; l < 2; ++l) {
        int naggs = NN * HH;
        zero_f32_kernel<<<(naggs + 255) / 256, 256, 0, stream>>>(agg, naggs);
        zero_f32_kernel<<<(NN + 255) / 256, 256, 0, stream>>>(nrm, NN);
        edge_attn_kernel<<<EE / 16, 256, 0, stream>>>(
            hb, edge_index, edge_feat, edge_times,
            te_w + l * TT, te_phi + l * TT,
            wq_p + (size_t)l * Q_TILES * 512, wq_b + l * HH,
            wk_p + (size_t)l * M_TILES * 512, wk_b + l * HH,
            wv_p + (size_t)l * M_TILES * 512, wv_b + l * HH,
            qte + l * 32, agg, nrm);
        node_update_kernel<<<NN / 8, 256, 0, stream>>>(
            agg, nrm, aln_w + l * HH, aln_b + l * HH,
            ln_w + l * HH, ln_b + l * HH, h, hb);
    }

    // 4) output projection + GELU
    out_proj_kernel<<<(NN + 31) / 32, 256, 0, stream>>>(hb, w_out_p, out_b, (float*)d_out);
}
